// Decoder_75505525064316
// MI455X (gfx1250) — matrix operand order repaired and checked by reading, not compiled, not run on hardware
//
#include <hip/hip_runtime.h>
#include <hip/hip_bf16.h>

// CDNA5 (gfx1250) fused decoder, v3:
//   y  = z @ |Wp|^T                 bf16 WMMA 16x16x32, K=64 (2 WMMA/tile)
//   h1 = relu(y*W1 + b1)            VALU, built directly in WMMA-A layout
//   h2 = relu(h1 @ W2[d]^T + b2)    bf16 WMMA, K=32, B reused over 4 n-subtiles
//   out= |h2 . W3[d] + b3|          split-butterfly ds_swizzle reduction
//
// v3 changes vs v2 (driven by valu/ds-heavy histogram):
//  * Column reduction: 4-stage split butterfly (values halve per stage):
//    15 ds_swizzle_b32 (immediate XOR masks) instead of 32 generic shfl_xor,
//    and each lane finishes holding exactly one output row -> single
//    LDS store from cols 0-7 (no 8x v_cmp_eq predication).
//  * All tile stores/loads use one base pointer + constant offsets so row
//    strides fold into DS immediate offsets (kills the v_mad per ds_store).

typedef __attribute__((ext_vector_type(16))) __bf16 v16bf;
typedef __attribute__((ext_vector_type(8)))  __bf16 v8bf;
typedef __attribute__((ext_vector_type(8)))  float  v8f;

#define LATENT 64
#define OUTF   128
#define H1D    32
#define H2D    32
#define NTILE  64          // rows per block
#define NSUB   4           // 16-row WMMA subtiles per block
#define NTHREADS 512       // 16 wave32 waves

// Padded LDS strides (64 banks x 4B)
#define ZB_STR  72         // z tile bf16 row stride; 144B, 16B-aligned rows
#define Y_STR   132        // y tile f32 row stride
#define WPT_STR 136        // |Wp|^T bf16 row stride; 272B, 16B-aligned rows

constexpr size_t SMEM_BYTES =
    (size_t)(NTILE * Y_STR + NTILE * OUTF + 2 * OUTF * H1D + 2 * OUTF * H2D + OUTF) * 4 +
    (size_t)(NTILE * ZB_STR + LATENT * WPT_STR) * 2;   // 159,232 B -> 2 blocks/WGP

// ds_swizzle_b32, group-of-32 XOR mode: new_lane = lane ^ m  (m<16 stays in half)
template <int IMM>
__device__ __forceinline__ float swz(float x) {
    return __int_as_float(__builtin_amdgcn_ds_swizzle(__float_as_int(x), IMM));
}
#define SWZ_X1 0x041F   // (1<<10)|0x1F
#define SWZ_X2 0x081F
#define SWZ_X4 0x101F
#define SWZ_X8 0x201F

// ---------------------------------------------------------------------------
// Prep: W2T[d][h][k] = bf16(W2[d][k][h])  (131072 elems, coalesced writes)
// ---------------------------------------------------------------------------
__global__ void prep_w2t(const float* __restrict__ W2, __bf16* __restrict__ W2T) {
    int j = blockIdx.x * 256 + threadIdx.x;            // [0, 128*32*32)
    int d = j >> 10, h = (j >> 5) & 31, k = j & 31;
    W2T[j] = (__bf16)W2[(d << 10) + (k << 5) + h];
}

// ---------------------------------------------------------------------------
// Fused decoder
// ---------------------------------------------------------------------------
__launch_bounds__(NTHREADS, 1)
__global__ void decoder_fused_wmma(const float* __restrict__ z,
                                   const float* __restrict__ Wp,
                                   const float* __restrict__ W1,
                                   const float* __restrict__ b1,
                                   const __bf16* __restrict__ W2T,
                                   const float* __restrict__ b2,
                                   const float* __restrict__ W3,
                                   const float* __restrict__ b3,
                                   float* __restrict__ out) {
    extern __shared__ unsigned char smem_raw[];
    float*  lds_y   = (float*)smem_raw;
    float*  lds_out = lds_y   + NTILE * Y_STR;
    float*  lds_w1  = lds_out + NTILE * OUTF;
    float*  lds_b1  = lds_w1  + OUTF * H1D;
    float*  lds_b2  = lds_b1  + OUTF * H1D;
    float*  lds_w3  = lds_b2  + OUTF * H2D;
    float*  lds_b3  = lds_w3  + OUTF * H2D;
    __bf16* lds_zb  = (__bf16*)(lds_b3 + OUTF);
    __bf16* lds_wpT = lds_zb + NTILE * ZB_STR;

    const int tid  = threadIdx.x;
    const int lane = tid & 31;
    const int wave = tid >> 5;
    const int n0   = blockIdx.x * NTILE;
    const int col  = lane & 15;      // C/D column == A row index
    const int hs   = lane >> 4;      // lane-half select
    const int k0   = hs << 3;        // A-fragment K base for this lane half

    // Warm L2 with the hot per-feature weights (global_prefetch_b8).
    __builtin_prefetch(W2T + ((tid * 128) & (OUTF * H2D * H1D - 1)), 0, 0);

    // ---- Cooperative fills ----------------------------------------------
    for (int i = tid; i < NTILE * LATENT; i += NTHREADS) {
        int m = i >> 6, k = i & (LATENT - 1);
        lds_zb[m * ZB_STR + k] = (__bf16)z[(size_t)(n0 + m) * LATENT + k];
    }
    for (int i = tid; i < OUTF * LATENT; i += NTHREADS) {
        int d = i >> 6, k = i & (LATENT - 1);
        lds_wpT[k * WPT_STR + d] = (__bf16)fabsf(Wp[i]);
    }
    for (int i = tid; i < OUTF * H1D; i += NTHREADS) {
        lds_w1[i] = W1[i];
        lds_b1[i] = b1[i];
        lds_b2[i] = b2[i];
        lds_w3[i] = W3[i];
    }
    if (tid < OUTF) lds_b3[tid] = b3[tid];
    __syncthreads();

    // ---- Phase A: y[64 x 128] = z_tile @ |Wp|^T -------------------------
    // 32 16x16 tiles; 16 waves x 2 tiles. A-element e -> K = k0+(e&7)+16*(e>>3).
    #pragma unroll
    for (int i = 0; i < 2; ++i) {
        const int t     = wave + 16 * i;     // 0..31
        const int sub   = t & 3;             // n-subtile
        const int dbase = (t >> 2) << 4;     // d-slab
        const __bf16* zrow = lds_zb + (sub * 16 + col) * ZB_STR;
        v8f acc = {};
        #pragma unroll
        for (int kc = 0; kc < 2; ++kc) {
            const int ka = kc * 32 + k0;
            v8bf alo = *(const v8bf*)(zrow + ka);
            v8bf ahi = *(const v8bf*)(zrow + ka + 16);
            v16bf a = __builtin_shufflevector(alo, ahi,
                0, 1, 2, 3, 4, 5, 6, 7, 8, 9, 10, 11, 12, 13, 14, 15);
            const __bf16* wcol = lds_wpT + (kc * 32) * WPT_STR + dbase + col;
            v16bf b;
            #pragma unroll
            for (int e = 0; e < 16; ++e)
                b[e] = wcol[(k0 + (e & 7) + 16 * (e >> 3)) * WPT_STR];
            acc = __builtin_amdgcn_wmma_f32_16x16x32_bf16(
                false, a, false, b, (short)0, acc, false, false);
        }
        float* yb = lds_y + (sub * 16 + 8 * hs) * Y_STR + dbase + col;
        #pragma unroll
        for (int r = 0; r < 8; ++r)
            yb[r * Y_STR] = acc[r];          // folds to immediate DS offsets
    }
    __syncthreads();

    // ---- Phase B: per-feature MLP; wave w owns d = w, w+16, ..., w+112 --
    const int b0s = col & 1, b1s = (col >> 1) & 1, b2s = (col >> 2) & 1;
    const int rme = 4 * b0s + 2 * b1s + b2s;           // row this lane finishes with
    float* outp_base = lds_out + (rme + 8 * hs) * OUTF;
    const float* yp = lds_y + col * Y_STR;

    for (int dd = 0; dd < 8; ++dd) {
        const int d = wave + 16 * dd;

        // B tiles: N=col (bt0) / 16+col (bt1), K = k0+(e&7)+16*(e>>3)
        const __bf16* w2p = W2T + (size_t)d * H2D * H1D + col;
        v16bf bt0, bt1;
        #pragma unroll
        for (int e = 0; e < 16; ++e) {
            const int kk = k0 + (e & 7) + 16 * (e >> 3);
            bt0[e] = w2p[kk * H2D];
            bt1[e] = w2p[kk * H2D + 16];
        }

        const v8f w1lo = *(const v8f*)(lds_w1 + d * H1D + k0);
        const v8f w1hi = *(const v8f*)(lds_w1 + d * H1D + k0 + 16);
        const v8f c1lo = *(const v8f*)(lds_b1 + d * H1D + k0);
        const v8f c1hi = *(const v8f*)(lds_b1 + d * H1D + k0 + 16);

        const float w3a = lds_w3[d * H2D + col];
        const float w3b = lds_w3[d * H2D + 16 + col];
        const float b2a = lds_b2[d * H2D + col];
        const float b2b = lds_b2[d * H2D + 16 + col];
        const float bb3 = lds_b3[d];

        const float* ypd  = yp + d;
        float*       outp = outp_base + d;

        #pragma unroll
        for (int sub = 0; sub < NSUB; ++sub) {
            const float yv = ypd[sub * 16 * Y_STR];    // y[m=col, d]

            // A = relu(y*W1 + b1) in WMMA-A layout (this IS layer 1)
            v16bf a;
            #pragma unroll
            for (int e = 0; e < 8; ++e) {
                a[e]     = (__bf16)fmaxf(fmaf(yv, w1lo[e], c1lo[e]), 0.0f);
                a[e + 8] = (__bf16)fmaxf(fmaf(yv, w1hi[e], c1hi[e]), 0.0f);
            }

            v8f c0 = {}, c1 = {};
            c0 = __builtin_amdgcn_wmma_f32_16x16x32_bf16(
                false, a, false, bt0, (short)0, c0, false, false);
            c1 = __builtin_amdgcn_wmma_f32_16x16x32_bf16(
                false, a, false, bt1, (short)0, c1, false, false);

            // q[r] = relu(h2+b2) dotted with W3 partials for this column
            float q[8];
            #pragma unroll
            for (int r = 0; r < 8; ++r) {
                const float h0 = fmaxf(c0[r] + b2a, 0.0f);
                const float h1 = fmaxf(c1[r] + b2b, 0.0f);
                q[r] = fmaf(h0, w3a, h1 * w3b);
            }

            // 4-stage split butterfly over the 16 columns (values halve/stage)
            float s[8];
            #pragma unroll
            for (int r = 0; r < 8; ++r) q[r] += swz<SWZ_X1>(q[r]);
            s[0] = b0s ? q[4] : q[0];
            s[1] = b0s ? q[5] : q[1];
            s[2] = b0s ? q[6] : q[2];
            s[3] = b0s ? q[7] : q[3];
            #pragma unroll
            for (int r = 0; r < 4; ++r) s[r] += swz<SWZ_X2>(s[r]);
            float u0 = b1s ? s[2] : s[0];
            float u1 = b1s ? s[3] : s[1];
            u0 += swz<SWZ_X4>(u0);
            u1 += swz<SWZ_X4>(u1);
            float v = b2s ? u1 : u0;
            v += swz<SWZ_X8>(v);
            // lane now holds x[sub*16 + rme + 8*hs, d]; cols 0-7 & 8-15 duplicate
            if ((col & 8) == 0)
                outp[sub * 16 * OUTF] = fabsf(v + bb3);
        }
    }
    __syncthreads();

    // ---- Coalesced tile store (32KB contiguous per block) ---------------
    for (int i = tid; i < NTILE * OUTF; i += NTHREADS)
        out[(size_t)n0 * OUTF + i] = lds_out[i];
}

extern "C" void kernel_launch(void* const* d_in, const int* in_sizes, int n_in,
                              void* d_out, int out_size, void* d_ws, size_t ws_size,
                              hipStream_t stream) {
    const float* z  = (const float*)d_in[0];
    const float* Wp = (const float*)d_in[1];
    const float* W1 = (const float*)d_in[2];
    const float* b1 = (const float*)d_in[3];
    const float* W2 = (const float*)d_in[4];
    const float* b2 = (const float*)d_in[5];
    const float* W3 = (const float*)d_in[6];
    const float* b3 = (const float*)d_in[7];
    float* out = (float*)d_out;

    __bf16* W2T = (__bf16*)d_ws;  // 128*32*32 bf16 = 256KB scratch

    prep_w2t<<<dim3((OUTF * H2D * H1D) / 256), 256, 0, stream>>>(W2, W2T);

    const int N = in_sizes[0] / LATENT;  // 65536
    decoder_fused_wmma<<<dim3(N / NTILE), NTHREADS, SMEM_BYTES, stream>>>(
        z, Wp, W1, b1, W2T, b2, W3, b3, out);
}
